// GraphCRF_72224170049681
// MI455X (gfx1250) — matrix operand order repaired and checked by reading, not compiled, not run on hardware
//
#include <hip/hip_runtime.h>
#include <hip/hip_bf16.h>

#define L_ 1024
#define N_ 16384
#define D_ 512
#define R_ 128
#define KC 128
#define PEN -10000.0f

typedef __attribute__((ext_vector_type(16))) __bf16 v16bf;
typedef __attribute__((ext_vector_type(8)))  float  v8f;
typedef __attribute__((ext_vector_type(4)))  int    v4i;

#if __has_builtin(__builtin_amdgcn_global_load_async_to_lds_b128) && \
    __has_builtin(__builtin_amdgcn_s_wait_asynccnt)
#define USE_ASYNC 1
#endif

#ifdef USE_ASYNC
typedef __attribute__((address_space(1))) v4i glob_v4i;
typedef __attribute__((address_space(3))) v4i lds_v4i;
// 16B global -> LDS async copy (ASYNCcnt-tracked, gfx1250)
__device__ __forceinline__ void stage16(unsigned short* l, const unsigned short* g) {
  __builtin_amdgcn_global_load_async_to_lds_b128(
      (glob_v4i*)g, (lds_v4i*)l, 0, 0);
}
__device__ __forceinline__ void stage_wait() { __builtin_amdgcn_s_wait_asynccnt(0); }
#else
__device__ __forceinline__ void stage16(unsigned short* l, const unsigned short* g) {
  *(uint4*)l = *(const uint4*)g;
}
__device__ __forceinline__ void stage_wait() {}
#endif

__device__ __forceinline__ unsigned short f32_to_bf16(float f) {
  unsigned u = __float_as_uint(f);
  unsigned rb = 0x7FFFu + ((u >> 16) & 1u);
  return (unsigned short)((u + rb) >> 16);
}

// A-matrix 16x32 bf16 fragment (ISA 7.12.2): lane<16 holds K{0..7,16..23},
// lane>=16 holds K{8..15,24..31}; M = lane%16.
template<int LDST>
__device__ __forceinline__ v16bf load_fragA(const unsigned short* base, int lane) {
  union { v16bf v; unsigned u[8]; } r;
  const int rrow = lane & 15;
  const int dof  = (lane < 16) ? 0 : 4;
  const unsigned short* p = base + rrow * LDST;
#pragma unroll
  for (int d = 0; d < 8; ++d) {
    const int kd = ((d < 4) ? d : d + 4) + dof;
    r.u[d] = *(const unsigned*)(p + kd * 2);
  }
  return r.v;
}

// B-matrix 32x16 bf16 fragment: lane<16 holds K{0..7,16..23}, lane>=16 holds K{8..15,24..31}; N = lane%16.
template<int LDST>
__device__ __forceinline__ v16bf load_fragB(const unsigned short* base, int lane) {
  union { v16bf v; unsigned u[8]; } r;
  const int rrow = lane & 15;
  const int dof  = (lane < 16) ? 0 : 4;
  const unsigned short* p = base + rrow * LDST;
#pragma unroll
  for (int d = 0; d < 8; ++d)
    r.u[d] = *(const unsigned*)(p + (((d < 4) ? d : d + 4) + dof) * 2);
  return r.v;
}

// ---------------------------------------------------------------------------
// bf16 conversion passes: Hbf = bf16(H); WbfT[n][k] = bf16(W[k][n])
// ---------------------------------------------------------------------------
__global__ __launch_bounds__(1024) void k_cvt_H(const float* __restrict__ H,
                                                unsigned short* __restrict__ Hbf) {
  const size_t i = (size_t)blockIdx.x * 1024 + threadIdx.x;
  if (i < (size_t)N_ * D_) Hbf[i] = f32_to_bf16(H[i]);
}

__global__ __launch_bounds__(1024) void k_cvt_WT(const float* __restrict__ W,
                                                 unsigned short* __restrict__ WbfT) {
  const int i = blockIdx.x * 1024 + threadIdx.x;
  if (i < D_ * D_) {
    const int k = i >> 9, n = i & (D_ - 1);
    WbfT[n * D_ + k] = f32_to_bf16(W[i]);
  }
}

// ---------------------------------------------------------------------------
// Kernel A: per-row top-128 via monotone-key radix binary search on a row
// cached in LDS; bitonic index sort; fused log_softmax of gathered logits.
// ---------------------------------------------------------------------------
__global__ __launch_bounds__(256) void k_topk(const float* __restrict__ U,
                                              int* __restrict__ cand,
                                              float* __restrict__ lsm_u) {
  __shared__ unsigned keys[N_];         // 64 KB row cache (320KB LDS/WGP on CDNA5)
  __shared__ int  red[256];
  __shared__ float fred[256];
  __shared__ int  hiList[R_];
  __shared__ int  eqList[R_ + 64];
  __shared__ int  cnts[2];
  __shared__ int  sel[R_];
  __shared__ float vals[R_];

  const int row = blockIdx.x;
  const int tid = threadIdx.x;
  const float* u = U + (size_t)row * N_;

  for (int i = tid; i < N_; i += 256) {
    unsigned b = __float_as_uint(u[i]);
    keys[i] = (b & 0x80000000u) ? ~b : (b | 0x80000000u);
  }
  __syncthreads();

  unsigned T = 0u;
  for (int bit = 31; bit >= 0; --bit) {
    const unsigned Tc = T | (1u << bit);
    int cnt = 0;
    for (int i = tid; i < N_; i += 256) cnt += (keys[i] >= Tc) ? 1 : 0;
    red[tid] = cnt; __syncthreads();
    for (int s = 128; s > 0; s >>= 1) { if (tid < s) red[tid] += red[tid + s]; __syncthreads(); }
    const int total = red[0];
    __syncthreads();
    if (total >= R_) T = Tc;
  }

  if (tid < 2) cnts[tid] = 0;
  __syncthreads();
  for (int i = tid; i < N_; i += 256) {
    const unsigned k = keys[i];
    if (k > T)       { int pos = atomicAdd(&cnts[0], 1); if (pos < R_)      hiList[pos] = i; }
    else if (k == T) { int pos = atomicAdd(&cnts[1], 1); if (pos < R_ + 64) eqList[pos] = i; }
  }
  __syncthreads();
  const int nHi = cnts[0];
  if (tid < R_) sel[tid] = (tid < nHi) ? hiList[tid] : eqList[tid - nHi];
  __syncthreads();

  for (int k = 2; k <= R_; k <<= 1) {
    for (int j = k >> 1; j > 0; j >>= 1) {
      if (tid < R_) {
        const int ixj = tid ^ j;
        if (ixj > tid) {
          const int a = sel[tid], b = sel[ixj];
          const bool up = ((tid & k) == 0);
          if ((a > b) == up) { sel[tid] = b; sel[ixj] = a; }
        }
      }
      __syncthreads();
    }
  }

  float v = 0.f;
  if (tid < R_) { v = u[sel[tid]]; vals[tid] = v; }
  __syncthreads();
  fred[tid] = (tid < R_) ? vals[tid] : -3.4e38f;
  __syncthreads();
  for (int s = 128; s > 0; s >>= 1) { if (tid < s) fred[tid] = fmaxf(fred[tid], fred[tid + s]); __syncthreads(); }
  const float mx = fred[0];
  __syncthreads();
  fred[tid] = (tid < R_) ? __expf(vals[tid] - mx) : 0.f;
  __syncthreads();
  for (int s = 128; s > 0; s >>= 1) { if (tid < s) fred[tid] += fred[tid + s]; __syncthreads(); }
  const float lse = mx + __logf(fred[0]);
  __syncthreads();
  if (tid < R_) {
    cand[row * R_ + tid]  = sel[tid];
    lsm_u[row * R_ + tid] = v - lse;
  }
}

// ---------------------------------------------------------------------------
// Kernel B: G = H @ W, bf16 WMMA from async-staged LDS tiles; writes G (fp32,
// for gold dot) and Gbf (bf16, for the batched pair GEMM).
// ---------------------------------------------------------------------------
__global__ __launch_bounds__(256) void k_gemm_HW(const unsigned short* __restrict__ Hbf,
                                                 const unsigned short* __restrict__ WbfT,
                                                 float* __restrict__ G,
                                                 unsigned short* __restrict__ Gbf) {
  __shared__ unsigned short As[128][KC + 8];   // H rows
  __shared__ unsigned short Bs[128][KC + 8];   // Bs[n][k] = W[k][n0+n]
  const int tid = threadIdx.x, lane = tid & 31, wave = tid >> 5;
  const int m0 = blockIdx.x * 128;
  const int n0 = blockIdx.y * 128;

  v8f acc[8] = {};
  for (int kc = 0; kc < D_; kc += KC) {
    for (int s = tid; s < 4096; s += 256) {
      const int mat = s >> 11, rem = s & 2047;
      const int r = rem >> 4, o = (rem & 15) * 8;
      if (mat == 0) stage16(&As[r][o], Hbf  + (size_t)(m0 + r) * D_ + kc + o);
      else          stage16(&Bs[r][o], WbfT + (size_t)(n0 + r) * D_ + kc + o);
    }
    stage_wait();
    __syncthreads();
#pragma unroll
    for (int kk = 0; kk < KC; kk += 32) {
      const v16bf a = load_fragA<KC + 8>(&As[wave * 16][kk], lane);
      v16bf b[8];
#pragma unroll
      for (int nt = 0; nt < 8; ++nt) b[nt] = load_fragB<KC + 8>(&Bs[nt * 16][kk], lane);
#pragma unroll
      for (int nt = 0; nt < 8; ++nt)
        acc[nt] = __builtin_amdgcn_wmma_f32_16x16x32_bf16(
            false, a, false, b[nt], (short)0, acc[nt], false, false);
    }
    __syncthreads();
  }
  const int nlo  = lane & 15;
  const int mofs = (lane < 16) ? 0 : 8;
#pragma unroll
  for (int nt = 0; nt < 8; ++nt)
#pragma unroll
    for (int v = 0; v < 8; ++v) {
      const int m = m0 + wave * 16 + v + mofs;
      const int n = n0 + nt * 16 + nlo;
      const float val = acc[nt][v];
      G  [(size_t)m * D_ + n] = val;
      Gbf[(size_t)m * D_ + n] = f32_to_bf16(val);
    }
}

// ---------------------------------------------------------------------------
// Kernel C: all 1023 pair matrices S[t-1] = G[prev] @ H[cur]^T + mask penalty.
// Gathered bf16 rows async-staged into LDS; one workgroup per step.
// ---------------------------------------------------------------------------
__global__ __launch_bounds__(256) void k_pair(const unsigned short* __restrict__ Gbf,
                                              const unsigned short* __restrict__ Hbf,
                                              const int* __restrict__ cand,
                                              const unsigned char* __restrict__ allowed,
                                              float* __restrict__ S) {
  __shared__ unsigned short As[128][KC + 8];   // G[prev] rows
  __shared__ unsigned short Bs[128][KC + 8];   // H[cur] rows (B[k][c] = H[cur[c]][k])
  __shared__ int prevIdx[R_], curIdx[R_];
  const int t   = blockIdx.x + 1;
  const int tid = threadIdx.x, lane = tid & 31, wave = tid >> 5;

  if (tid < R_)          prevIdx[tid]     = cand[(t - 1) * R_ + tid];
  else if (tid < 2 * R_) curIdx[tid - R_] = cand[t * R_ + (tid - R_)];
  __syncthreads();

  v8f acc[8] = {};
  for (int kc = 0; kc < D_; kc += KC) {
    for (int s = tid; s < 4096; s += 256) {
      const int mat = s >> 11, rem = s & 2047;
      const int r = rem >> 4, o = (rem & 15) * 8;
      if (mat == 0) stage16(&As[r][o], Gbf + (size_t)prevIdx[r] * D_ + kc + o);
      else          stage16(&Bs[r][o], Hbf + (size_t)curIdx[r]  * D_ + kc + o);
    }
    stage_wait();
    __syncthreads();
#pragma unroll
    for (int kk = 0; kk < KC; kk += 32) {
      const v16bf a = load_fragA<KC + 8>(&As[wave * 16][kk], lane);
      v16bf b[8];
#pragma unroll
      for (int nt = 0; nt < 8; ++nt) b[nt] = load_fragB<KC + 8>(&Bs[nt * 16][kk], lane);
#pragma unroll
      for (int nt = 0; nt < 8; ++nt)
        acc[nt] = __builtin_amdgcn_wmma_f32_16x16x32_bf16(
            false, a, false, b[nt], (short)0, acc[nt], false, false);
    }
    __syncthreads();
  }

  const int clo  = lane & 15;
  const int mofs = (lane < 16) ? 0 : 8;
  float* Sout = S + (size_t)(t - 1) * R_ * R_;
#pragma unroll
  for (int nt = 0; nt < 8; ++nt)
#pragma unroll
    for (int v = 0; v < 8; ++v) {
      const int p = wave * 16 + v + mofs;
      const int c = nt * 16 + clo;
      const bool ok = allowed[(size_t)curIdx[c] * N_ + prevIdx[p]] != 0;
      Sout[p * R_ + c] = acc[nt][v] + (ok ? 0.f : PEN);
    }
}

// ---------------------------------------------------------------------------
// Kernel E: gold-path score
// ---------------------------------------------------------------------------
__global__ __launch_bounds__(256) void k_gold(const float* __restrict__ U,
                                              const float* __restrict__ G,
                                              const float* __restrict__ H,
                                              const int* __restrict__ gold,
                                              const unsigned char* __restrict__ allowed,
                                              float* __restrict__ goldScore) {
  __shared__ float red[256];
  const int tid = threadIdx.x;
  float s = 0.f;
  for (int t = tid; t < L_; t += 256) s += U[(size_t)t * N_ + gold[t]];
  for (int t = tid; t < L_ - 1; t += 256) {
    const int gp = gold[t], gc = gold[t + 1];
    const float* gr = G + (size_t)gp * D_;
    const float* hr = H + (size_t)gc * D_;
    float d = 0.f;
    for (int k = 0; k < D_; ++k) d = fmaf(gr[k], hr[k], d);
    s += (allowed[(size_t)gc * N_ + gp] != 0) ? d : PEN;
  }
  red[tid] = s; __syncthreads();
  for (int st = 128; st > 0; st >>= 1) { if (tid < st) red[tid] += red[tid + st]; __syncthreads(); }
  if (tid == 0) goldScore[0] = red[0];
}

// ---------------------------------------------------------------------------
// Kernel D: sequential alpha recursion (single WG, 512 threads = 4 partial
// reducers per column), then logZ and the final scalar.
// ---------------------------------------------------------------------------
__global__ __launch_bounds__(512) void k_alpha(const float* __restrict__ S,
                                               const float* __restrict__ lsm_u,
                                               const float* __restrict__ goldScore,
                                               float* __restrict__ out) {
  __shared__ float alpha[R_];
  __shared__ float tmpM[512], tmpS[512];
  const int tid = threadIdx.x;
  const int c = tid & 127;
  const int h = tid >> 7;            // 0..3

  if (tid < R_) alpha[tid] = lsm_u[tid];
  __syncthreads();

  for (int t = 1; t < L_; ++t) {
    const float* Sc = S + (size_t)(t - 1) * R_ * R_ + c;
    float m = -3.4e38f, sm = 0.f;
    const int p0 = h * 32;
#pragma unroll 4
    for (int pp = 0; pp < 32; ++pp) {
      const int p = p0 + pp;
      const float x = alpha[p] + Sc[(size_t)p * R_];
      if (x > m) { sm = sm * __expf(m - x) + 1.f; m = x; }
      else       { sm += __expf(x - m); }
    }
    tmpM[tid] = m; tmpS[tid] = sm;
    __syncthreads();
    float na = 0.f;
    if (h == 0) {
      float M = m;
#pragma unroll
      for (int q = 1; q < 4; ++q) M = fmaxf(M, tmpM[c + q * 128]);
      float Ssum = sm * __expf(m - M);
#pragma unroll
      for (int q = 1; q < 4; ++q) Ssum += tmpS[c + q * 128] * __expf(tmpM[c + q * 128] - M);
      na = M + __logf(Ssum) + lsm_u[t * R_ + c];
    }
    __syncthreads();
    if (h == 0) alpha[c] = na;
    __syncthreads();
  }

  tmpM[tid] = (tid < R_) ? alpha[tid] : -3.4e38f;
  __syncthreads();
  for (int s = 256; s > 0; s >>= 1) { if (tid < s) tmpM[tid] = fmaxf(tmpM[tid], tmpM[tid + s]); __syncthreads(); }
  const float mx = tmpM[0];
  __syncthreads();
  tmpS[tid] = (tid < R_) ? __expf(alpha[tid] - mx) : 0.f;
  __syncthreads();
  for (int s = 256; s > 0; s >>= 1) { if (tid < s) tmpS[tid] += tmpS[tid + s]; __syncthreads(); }
  if (tid == 0) out[0] = (mx + __logf(tmpS[0])) - goldScore[0];
}

// ---------------------------------------------------------------------------
extern "C" void kernel_launch(void* const* d_in, const int* in_sizes, int n_in,
                              void* d_out, int out_size, void* d_ws, size_t ws_size,
                              hipStream_t stream) {
  (void)in_sizes; (void)n_in; (void)out_size; (void)ws_size;
  const float*         unary   = (const float*)d_in[0];
  const float*         H       = (const float*)d_in[1];
  const float*         W       = (const float*)d_in[2];
  const int*           gold    = (const int*)d_in[3];
  const unsigned char* allowed = (const unsigned char*)d_in[4];
  // d_in[5] = top_r (constant 128, baked in as R_)

  char* ws = (char*)d_ws;
  float*          G         = (float*)ws;          ws += (size_t)N_ * D_ * sizeof(float);
  float*          S         = (float*)ws;          ws += (size_t)(L_ - 1) * R_ * R_ * sizeof(float);
  unsigned short* Hbf       = (unsigned short*)ws; ws += (size_t)N_ * D_ * sizeof(unsigned short);
  unsigned short* Gbf       = (unsigned short*)ws; ws += (size_t)N_ * D_ * sizeof(unsigned short);
  unsigned short* WbfT      = (unsigned short*)ws; ws += (size_t)D_ * D_ * sizeof(unsigned short);
  int*            cand      = (int*)ws;            ws += (size_t)L_ * R_ * sizeof(int);
  float*          lsm       = (float*)ws;          ws += (size_t)L_ * R_ * sizeof(float);
  float*          goldScore = (float*)ws;          ws += 256;

  k_cvt_H  <<<(N_ * D_ + 1023) / 1024, 1024, 0, stream>>>(H, Hbf);
  k_cvt_WT <<<(D_ * D_ + 1023) / 1024, 1024, 0, stream>>>(W, WbfT);
  k_topk   <<<L_, 256, 0, stream>>>(unary, cand, lsm);
  k_gemm_HW<<<dim3(N_ / 128, D_ / 128), 256, 0, stream>>>(Hbf, WbfT, G, Gbf);
  k_pair   <<<L_ - 1, 256, 0, stream>>>(Gbf, Hbf, cand, allowed, S);
  k_gold   <<<1, 256, 0, stream>>>(unary, G, H, gold, allowed, goldScore);
  k_alpha  <<<1, 512, 0, stream>>>(S, lsm, goldScore, (float*)d_out);
}
